// JiT_50568944943390
// MI455X (gfx1250) — matrix operand order repaired and checked by reading, not compiled, not run on hardware
//
#include <hip/hip_runtime.h>
#include <math.h>

// ---------------------------------------------------------------------------
// DiT forward for MI455X (gfx1250, wave32).
// All GEMMs via v_wmma_f32_16x16x32_f16 (f32 accumulate). f32 weights stream
// from HBM once per layer (74MB/layer fits the 192MB L2), converted to f16
// during LDS staging. Double-buffered LDS keeps global loads in flight
// during the WMMA burst (the kernel is HBM/L2 bandwidth bound).
// ---------------------------------------------------------------------------

typedef _Float16 half_t;
typedef __attribute__((ext_vector_type(16))) _Float16 v16h;
typedef __attribute__((ext_vector_type(8)))  _Float16 v8h;
typedef __attribute__((ext_vector_type(8)))  float    v8f;

#define BMT 128
#define BNT 128
#define BKT 32
#define LDT 40   // padded LDS row stride (halves): 80B rows, conflict-free

// --- fragment builders (16B ds_load_b128 chunks out of LDS) ----------------
__device__ __forceinline__ v16h frag_a(const half_t* tile, int row, int hs) {
  const half_t* p = tile + row * LDT;
  v8h c0 = *(const v8h*)(p + hs * 8);
  v8h c1 = *(const v8h*)(p + 16 + hs * 8);
  union { v16h v; v8h h[2]; } u; u.h[0] = c0; u.h[1] = c1; return u.v;
}
__device__ __forceinline__ v16h frag_b(const half_t* tile, int row, int hs) {
  const half_t* p = tile + row * LDT;
  v8h c0 = *(const v8h*)(p + hs * 8);
  v8h c1 = *(const v8h*)(p + 16 + hs * 8);
  union { v16h v; v8h h[2]; } u; u.h[0] = c0; u.h[1] = c1; return u.v;
}

// ---------------------------------------------------------------------------
// Generic batched GEMM:  out[b] = act(X[b] @ W[b]^T * alpha + bias) * gate + resid
//   X: [M,K] row stride ldx, batch stride bsx   (f32)
//   W: [N,K] row stride ldw, batch stride bsw   (f32, "NT" layout)
//   gate index = (row>>8)*gate_ld + col
// All row strides used by the model are even, so float2 (8B) staging loads
// are always aligned (incl. the MH=2730 strides).
// ---------------------------------------------------------------------------
__global__ __launch_bounds__(256) void gemm_wmma_kernel(
    const float* __restrict__ X, long long bsx, int ldx,
    const float* __restrict__ W, long long bsw, int ldw,
    const float* __restrict__ bias,
    const float* __restrict__ resid,
    const float* __restrict__ gate, int gate_ld,
    float* __restrict__ out, long long bso, int ldo,
    int M, int N, int K, float alpha, int act)
{
  __shared__ alignas(16) half_t Xs[2][BMT * LDT];
  __shared__ alignas(16) half_t Ws[2][BNT * LDT];

  const int bz = blockIdx.z;
  X += (long long)bz * bsx;
  W += (long long)bz * bsw;
  out += (long long)bz * bso;
  if (resid) resid += (long long)bz * bso;

  const int n0 = blockIdx.x * BNT, m0 = blockIdx.y * BMT;
  const int tid = threadIdx.x;
  const int wave = tid >> 5, lane = tid & 31;
  const int wm = (wave >> 2) * 64;   // 2 wave-rows of 64
  const int wn = (wave & 3) * 32;    // 4 wave-cols of 32
  const int hs = lane >> 4, r16 = lane & 15;

  const bool intX = (m0 + BMT <= M);   // X tile fully interior in M
  const bool intW = (n0 + BNT <= N);   // W tile fully interior in N

  const v8f vzero = {0.f,0.f,0.f,0.f,0.f,0.f,0.f,0.f};
  v8f acc[4][2];
  for (int mt = 0; mt < 4; ++mt) for (int nt = 0; nt < 2; ++nt) acc[mt][nt] = vzero;

  const int KT = (K + BKT - 1) / BKT;

  float2 xr[8], wr[8];   // staged registers (global -> reg -> LDS)

  // Load one 128x32 K-tile of X and W into registers (branchless edges:
  // clamped addresses + select-zero, so loads batch and wait once).
  auto load_tile = [&](int k0) {
    const bool fk = (k0 + BKT <= K);
#pragma unroll
    for (int i = 0; i < 8; ++i) {
      int pos = i * 256 + tid;
      int r = pos >> 4, kq = pos & 15;
      if (intX && fk) {
        xr[i] = *(const float2*)(X + (long long)(m0 + r) * ldx + k0 + kq * 2);
      } else {
        int row = m0 + r;
        int cr = row < M ? row : M - 1;
        int g0 = k0 + kq * 2, g1 = g0 + 1;
        int c0 = g0 < K ? g0 : K - 1, c1 = g1 < K ? g1 : K - 1;
        float a0 = X[(long long)cr * ldx + c0];
        float a1 = X[(long long)cr * ldx + c1];
        xr[i].x = (row < M && g0 < K) ? a0 : 0.f;
        xr[i].y = (row < M && g1 < K) ? a1 : 0.f;
      }
      if (intW && fk) {
        wr[i] = *(const float2*)(W + (long long)(n0 + r) * ldw + k0 + kq * 2);
      } else {
        int row = n0 + r;
        int cr = row < N ? row : N - 1;
        int g0 = k0 + kq * 2, g1 = g0 + 1;
        int c0 = g0 < K ? g0 : K - 1, c1 = g1 < K ? g1 : K - 1;
        float a0 = W[(long long)cr * ldw + c0];
        float a1 = W[(long long)cr * ldw + c1];
        wr[i].x = (row < N && g0 < K) ? a0 : 0.f;
        wr[i].y = (row < N && g1 < K) ? a1 : 0.f;
      }
    }
  };
  // Convert to f16 and store staged registers into LDS buffer `buf`.
  auto store_tile = [&](int buf) {
#pragma unroll
    for (int i = 0; i < 8; ++i) {
      int pos = i * 256 + tid;
      int r = pos >> 4, kq = pos & 15;
      union { half_t h[2]; unsigned u; } px, pw;
      px.h[0] = (half_t)xr[i].x; px.h[1] = (half_t)xr[i].y;
      pw.h[0] = (half_t)wr[i].x; pw.h[1] = (half_t)wr[i].y;
      *(unsigned*)(&Xs[buf][r * LDT + kq * 2]) = px.u;
      *(unsigned*)(&Ws[buf][r * LDT + kq * 2]) = pw.u;
    }
  };

  load_tile(0);
  store_tile(0);
  __syncthreads();

  for (int kt = 0; kt < KT; ++kt) {
    const int cur = kt & 1;
    // issue next tile's global loads before the WMMA burst (double buffer)
    if (kt + 1 < KT) {
      load_tile((kt + 1) * BKT);
      int pk = (kt + 2) * BKT;               // prefetch two tiles ahead
      if (pk < K) {
        int rr = tid & 127;
        if (tid < 128) { if (m0 + rr < M) __builtin_prefetch(&X[(long long)(m0 + rr) * ldx + pk], 0, 1); }
        else           { if (n0 + rr < N) __builtin_prefetch(&W[(long long)(n0 + rr) * ldw + pk], 0, 1); }
      }
    }
    // hoist all fragment ds_loads, then the 8 WMMAs
    v16h bf0 = frag_b(Ws[cur], wn + r16, hs);
    v16h bf1 = frag_b(Ws[cur], wn + 16 + r16, hs);
    v16h af[4];
#pragma unroll
    for (int mt = 0; mt < 4; ++mt) af[mt] = frag_a(Xs[cur], wm + mt * 16 + r16, hs);
#pragma unroll
    for (int mt = 0; mt < 4; ++mt) {
      acc[mt][0] = __builtin_amdgcn_wmma_f32_16x16x32_f16(false, af[mt], false, bf0,
                      (short)0, acc[mt][0], false, false);
      acc[mt][1] = __builtin_amdgcn_wmma_f32_16x16x32_f16(false, af[mt], false, bf1,
                      (short)0, acc[mt][1], false, false);
    }
    if (kt + 1 < KT) store_tile(1 - cur);
    __syncthreads();
  }

  // epilogue: C/D layout -> (row = i + 8*hs, col = r16) per 16x16 tile
  for (int mt = 0; mt < 4; ++mt) {
    for (int nt = 0; nt < 2; ++nt) {
      int col = n0 + wn + nt * 16 + r16;
      for (int i = 0; i < 8; ++i) {
        int row = m0 + wm + mt * 16 + hs * 8 + i;
        if (row < M && col < N) {
          float v = acc[mt][nt][i] * alpha;
          if (bias)  v += bias[col];
          if (act == 1) v = v * (1.f / (1.f + __expf(-v)));   // silu
          if (gate)  v *= gate[(long long)(row >> 8) * gate_ld + col];
          if (resid) v += resid[(long long)row * ldo + col];
          out[(long long)row * ldo + col] = v;
        }
      }
    }
  }
}

// --------------------------- elementwise kernels ---------------------------
__global__ void patchify_kernel(const float* __restrict__ x, float* __restrict__ xp) {
  int i = blockIdx.x * 256 + threadIdx.x;
  if (i >= 4096 * 768) return;
  int row = i / 768, j = i - row * 768;
  int b = row >> 8, n = row & 255;
  int c = j >> 8, p = j & 255;
  int py = p >> 4, px = p & 15;
  int gy = n >> 4, gx = n & 15;
  xp[i] = x[(((long long)(b * 3 + c) * 256) + gy * 16 + py) * 256 + gx * 16 + px];
}

__global__ void pos_add_kernel(float* __restrict__ h, const float* __restrict__ pos) {
  int i = blockIdx.x * 256 + threadIdx.x;
  if (i >= 4096 * 1024) return;
  h[i] += pos[i & (256 * 1024 - 1)];
}

__global__ void temb_freq_kernel(const float* __restrict__ t, float* __restrict__ o) {
  int i = blockIdx.x * 256 + threadIdx.x;
  if (i >= 16 * 256) return;
  int b = i >> 8, j = i & 255;
  float fr = __expf(-logf(10000.f) * (float)(j & 127) / 128.f);
  float a = t[b] * fr;
  o[i] = (j < 128) ? cosf(a) : sinf(a);
}

__global__ void c_silu_kernel(const float* __restrict__ temb, const float* __restrict__ ytab,
                              const int* __restrict__ y, float* __restrict__ sc) {
  int i = blockIdx.x * 256 + threadIdx.x;
  if (i >= 16 * 1024) return;
  int b = i >> 10, d = i & 1023;
  float c = temb[i] + ytab[(long long)y[b] * 1024 + d];
  sc[i] = c * (1.f / (1.f + __expf(-c)));
}

__global__ void rope_tables_kernel(float* __restrict__ rc, float* __restrict__ rs) {
  int i = blockIdx.x * 256 + threadIdx.x;
  if (i >= 256 * 64) return;
  int n = i >> 6, d = i & 63;
  int ga = n >> 4, gb = n & 15;
  int fi = (d < 32) ? (d >> 1) : ((d - 32) >> 1);
  float tt = (float)((d < 32) ? ga : gb);
  float fr = __expf(-logf(10000.f) * (float)(2 * fi) / 32.f);
  float ang = tt * fr;
  rc[i] = cosf(ang);
  rs[i] = sinf(ang);
}

// rms(x)*w*(1+scale[b])+shift[b]  over D=1024, one block per token row
__global__ __launch_bounds__(256) void rms_mod_kernel(
    const float* __restrict__ h, const float* __restrict__ w,
    const float* __restrict__ scale, const float* __restrict__ shift,
    int cond_ld, float* __restrict__ out)
{
  int row = blockIdx.x;
  int b = row >> 8;
  const float* hr = h + (long long)row * 1024;
  float vals[4]; float ss = 0.f;
  for (int j = 0; j < 4; ++j) {
    vals[j] = hr[threadIdx.x + j * 256];
    ss += vals[j] * vals[j];
  }
  for (int m = 1; m < 32; m <<= 1) ss += __shfl_xor(ss, m, 32);
  __shared__ float part[8];
  int wid = threadIdx.x >> 5, lane = threadIdx.x & 31;
  if (lane == 0) part[wid] = ss;
  __syncthreads();
  float tot = 0.f;
  for (int i = 0; i < 8; ++i) tot += part[i];
  float inv = rsqrtf(tot / 1024.f + 1e-6f);
  for (int j = 0; j < 4; ++j) {
    int col = threadIdx.x + j * 256;
    out[(long long)row * 1024 + col] =
        vals[j] * inv * w[col] * (1.f + scale[(long long)b * cond_ld + col]) +
        shift[(long long)b * cond_ld + col];
  }
}

// per-(b,h): q/k RMSNorm + RoPE into (BH,N,64); v transposed into (BH,64,N)
__global__ __launch_bounds__(256) void qk_prep_kernel(
    const float* __restrict__ qkv, const float* __restrict__ qw,
    const float* __restrict__ kw, const float* __restrict__ rc,
    const float* __restrict__ rs, float* __restrict__ qb,
    float* __restrict__ kb, float* __restrict__ vT)
{
  int bh = blockIdx.x;
  int b = bh >> 4, hh = bh & 15;
  int wid = threadIdx.x >> 5, lane = threadIdx.x & 31;
  int d = lane * 2;
  float qw0 = qw[d], qw1 = qw[d + 1], kw0 = kw[d], kw1 = kw[d + 1];
  for (int n = wid; n < 256; n += 8) {
    const float* base = qkv + (long long)(b * 256 + n) * 3072 + hh * 64 + d;
    float q0 = base[0],    q1 = base[1];
    float k0 = base[1024], k1 = base[1025];
    float v0 = base[2048], v1 = base[2049];
    float sq = q0 * q0 + q1 * q1, sk = k0 * k0 + k1 * k1;
    for (int m = 1; m < 32; m <<= 1) { sq += __shfl_xor(sq, m, 32); sk += __shfl_xor(sk, m, 32); }
    float iq = rsqrtf(sq / 64.f + 1e-6f), ik = rsqrtf(sk / 64.f + 1e-6f);
    q0 *= iq * qw0; q1 *= iq * qw1; k0 *= ik * kw0; k1 *= ik * kw1;
    float c0 = rc[n * 64 + d], c1 = rc[n * 64 + d + 1];
    float s0 = rs[n * 64 + d], s1 = rs[n * 64 + d + 1];
    float Q0 = q0 * c0 - q1 * s0, Q1 = q1 * c1 + q0 * s1;   // q*cos + rot_half(q)*sin
    float K0 = k0 * c0 - k1 * s0, K1 = k1 * c1 + k0 * s1;
    long long ob = (long long)bh * 16384 + n * 64 + d;
    qb[ob] = Q0; qb[ob + 1] = Q1;
    kb[ob] = K0; kb[ob + 1] = K1;
    vT[(long long)bh * 16384 + d * 256 + n] = v0;
    vT[(long long)bh * 16384 + (d + 1) * 256 + n] = v1;
  }
}

// row softmax over 256 cols, one wave per row
__global__ __launch_bounds__(256) void softmax_kernel(float* __restrict__ S) {
  int row = blockIdx.x * 8 + (threadIdx.x >> 5);
  int lane = threadIdx.x & 31;
  float* r = S + (long long)row * 256;
  float v[8]; float mx = -1e30f;
  for (int j = 0; j < 8; ++j) { v[j] = r[lane + j * 32]; mx = fmaxf(mx, v[j]); }
  for (int m = 1; m < 32; m <<= 1) mx = fmaxf(mx, __shfl_xor(mx, m, 32));
  float s = 0.f;
  for (int j = 0; j < 8; ++j) { v[j] = __expf(v[j] - mx); s += v[j]; }
  for (int m = 1; m < 32; m <<= 1) s += __shfl_xor(s, m, 32);
  float inv = 1.f / s;
  for (int j = 0; j < 8; ++j) r[lane + j * 32] = v[j] * inv;
}

// (b,h,n,d) -> (b,n,h*64+d)
__global__ void repack_o_kernel(const float* __restrict__ o1, float* __restrict__ o2) {
  int i = blockIdx.x * 256 + threadIdx.x;
  if (i >= 4096 * 1024) return;
  int row = i >> 10, col = i & 1023;
  int b = row >> 8, n = row & 255;
  int hh = col >> 6, d = col & 63;
  o2[i] = o1[((long long)((b * 16 + hh) * 256 + n)) * 64 + d];
}

__global__ void silu_gate_kernel(const float* __restrict__ x12, float* __restrict__ g) {
  int i = blockIdx.x * 256 + threadIdx.x;
  if (i >= 4096 * 2730) return;
  int row = i / 2730, j = i - row * 2730;
  float a = x12[(long long)row * 5460 + j];
  float b = x12[(long long)row * 5460 + 2730 + j];
  g[i] = a * (1.f / (1.f + __expf(-a))) * b;
}

__global__ void unpatchify_kernel(const float* __restrict__ flin, float* __restrict__ o) {
  int i = blockIdx.x * 256 + threadIdx.x;
  if (i >= 16 * 3 * 65536) return;
  int b = i / 196608; int r1 = i - b * 196608;
  int c = r1 / 65536; int p = r1 - c * 65536;
  int rr = p >> 8, cc = p & 255;
  int gy = rr >> 4, py = rr & 15, gx = cc >> 4, px = cc & 15;
  int n = gy * 16 + gx;
  int j = py * 48 + px * 3 + c;
  o[i] = flin[(long long)(b * 256 + n) * 768 + j];
}

// ---------------------------------------------------------------------------
extern "C" void kernel_launch(void* const* d_in, const int* in_sizes, int n_in,
                              void* d_out, int out_size, void* d_ws, size_t ws_size,
                              hipStream_t stream) {
  (void)in_sizes; (void)n_in; (void)out_size; (void)ws_size;
  const float* x    = (const float*)d_in[0];
  const float* t_in = (const float*)d_in[1];
  const int*   y_in = (const int*)d_in[2];
  const float* pos  = (const float*)d_in[3];
  const float* pw1  = (const float*)d_in[4];
  const float* pw2  = (const float*)d_in[5];
  const float* pb2  = (const float*)d_in[6];
  const float* tw1  = (const float*)d_in[7];
  const float* tb1  = (const float*)d_in[8];
  const float* tw2  = (const float*)d_in[9];
  const float* tb2  = (const float*)d_in[10];
  const float* ytab = (const float*)d_in[11];
  const float* n1w  = (const float*)d_in[12];
  const float* qkvw = (const float*)d_in[13];
  const float* qkvb = (const float*)d_in[14];
  const float* qnw  = (const float*)d_in[15];
  const float* knw  = (const float*)d_in[16];
  const float* pjw  = (const float*)d_in[17];
  const float* pjb  = (const float*)d_in[18];
  const float* n2w  = (const float*)d_in[19];
  const float* w12w = (const float*)d_in[20];
  const float* w12b = (const float*)d_in[21];
  const float* w3w  = (const float*)d_in[22];
  const float* w3b  = (const float*)d_in[23];
  const float* adaw = (const float*)d_in[24];
  const float* adab = (const float*)d_in[25];
  const float* fnw  = (const float*)d_in[26];
  const float* flw  = (const float*)d_in[27];
  const float* flb  = (const float*)d_in[28];
  const float* fdw  = (const float*)d_in[29];
  const float* fdb  = (const float*)d_in[30];
  float* outp = (float*)d_out;

  // ----- workspace layout (floats) -----
  float* w = (float*)d_ws;
  size_t off = 0;
  auto take = [&](size_t n) { float* p = w + off; off += n; return p; };
  float* hbuf   = take(4194304);      // (B*N, D)
  float* xmbuf  = take(4194304);
  float* qkvbuf = take(12582912);     // (B*N, 3D)
  float* qb     = take(4194304);      // (BH, N, 64)
  float* kb     = take(4194304);
  float* vT     = take(4194304);      // (BH, 64, N)
  float* o2     = take(4194304);      // (B*N, D)
  float* U      = take(33546240);     // phase-aliased big region
  float* tembin = take(4096);
  float* th     = take(16384);
  float* temb   = take(16384);
  float* siluc  = take(16384);
  float* ada    = take(98304);        // (B, 6D)
  float* fada   = take(32768);        // (B, 2D)
  float* rc     = take(16384);
  float* rs     = take(16384);
  // aliases (disjoint in time within a layer):
  float* S    = U;                    // (BH, 256, 256)
  float* o1   = U + 16777216;         // (BH, 256, 64)
  float* x12  = U;                    // (B*N, 5460)
  float* gbuf = U + 22364160;         // (B*N, 2730)
  float* xp   = U;                    // (B*N, 768) pre-layers only
  float* tmp  = U + 3145728;          // (B*N, 128)
  float* flin = U;                    // (B*N, 768) post-layers only

  auto gemm = [&](const float* Xp, long long bsx, int ldx,
                  const float* Wp, long long bsw, int ldw,
                  const float* bp, const float* rp,
                  const float* gp, int gld,
                  float* op, long long bso, int ldo,
                  int M, int N, int K, float alpha, int act, int batch) {
    dim3 grid((N + BNT - 1) / BNT, (M + BMT - 1) / BMT, batch);
    gemm_wmma_kernel<<<grid, 256, 0, stream>>>(Xp, bsx, ldx, Wp, bsw, ldw, bp, rp,
                                               gp, gld, op, bso, ldo, M, N, K, alpha, act);
  };
  auto blocks = [](int total) { return dim3((total + 255) / 256); };

  // ----- patch embed -----
  patchify_kernel<<<blocks(4096 * 768), 256, 0, stream>>>(x, xp);
  gemm(xp, 0, 768, pw1, 0, 768, nullptr, nullptr, nullptr, 0,
       tmp, 0, 128, 4096, 128, 768, 1.f, 0, 1);
  gemm(tmp, 0, 128, pw2, 0, 128, pb2, nullptr, nullptr, 0,
       hbuf, 0, 1024, 4096, 1024, 128, 1.f, 0, 1);
  pos_add_kernel<<<blocks(4096 * 1024), 256, 0, stream>>>(hbuf, pos);

  // ----- conditioning -----
  temb_freq_kernel<<<blocks(16 * 256), 256, 0, stream>>>(t_in, tembin);
  gemm(tembin, 0, 256, tw1, 0, 256, tb1, nullptr, nullptr, 0,
       th, 0, 1024, 16, 1024, 256, 1.f, 1 /*silu*/, 1);
  gemm(th, 0, 1024, tw2, 0, 1024, tb2, nullptr, nullptr, 0,
       temb, 0, 1024, 16, 1024, 1024, 1.f, 0, 1);
  c_silu_kernel<<<blocks(16 * 1024), 256, 0, stream>>>(temb, ytab, y_in, siluc);
  rope_tables_kernel<<<blocks(256 * 64), 256, 0, stream>>>(rc, rs);

  // ----- transformer layers -----
  for (int i = 0; i < 12; ++i) {
    gemm(siluc, 0, 1024, adaw + (long long)i * 6144 * 1024, 0, 1024,
         adab + (long long)i * 6144, nullptr, nullptr, 0,
         ada, 0, 6144, 16, 6144, 1024, 1.f, 0, 1);

    rms_mod_kernel<<<4096, 256, 0, stream>>>(hbuf, n1w + i * 1024,
                                             ada + 1024, ada, 6144, xmbuf);
    gemm(xmbuf, 0, 1024, qkvw + (long long)i * 3072 * 1024, 0, 1024,
         qkvb + (long long)i * 3072, nullptr, nullptr, 0,
         qkvbuf, 0, 3072, 4096, 3072, 1024, 1.f, 0, 1);

    qk_prep_kernel<<<256, 256, 0, stream>>>(qkvbuf, qnw + i * 64, knw + i * 64,
                                            rc, rs, qb, kb, vT);
    // scores = softmax(Q K^T / 8)
    gemm(qb, 16384, 64, kb, 16384, 64, nullptr, nullptr, nullptr, 0,
         S, 65536, 256, 256, 256, 64, 0.125f, 0, 256);
    softmax_kernel<<<8192, 256, 0, stream>>>(S);
    gemm(S, 65536, 256, vT, 16384, 256, nullptr, nullptr, nullptr, 0,
         o1, 16384, 64, 256, 64, 256, 1.f, 0, 256);
    repack_o_kernel<<<blocks(4096 * 1024), 256, 0, stream>>>(o1, o2);
    // h += gm * (o @ proj^T + b)
    gemm(o2, 0, 1024, pjw + (long long)i * 1024 * 1024, 0, 1024,
         pjb + (long long)i * 1024, hbuf, ada + 2048, 6144,
         hbuf, 0, 1024, 4096, 1024, 1024, 1.f, 0, 1);

    rms_mod_kernel<<<4096, 256, 0, stream>>>(hbuf, n2w + i * 1024,
                                             ada + 4096, ada + 3072, 6144, xmbuf);
    gemm(xmbuf, 0, 1024, w12w + (long long)i * 5460 * 1024, 0, 1024,
         w12b + (long long)i * 5460, nullptr, nullptr, 0,
         x12, 0, 5460, 4096, 5460, 1024, 1.f, 0, 1);
    silu_gate_kernel<<<blocks(4096 * 2730), 256, 0, stream>>>(x12, gbuf);
    // h += gp * (g @ w3^T + b)
    gemm(gbuf, 0, 2730, w3w + (long long)i * 1024 * 2730, 0, 2730,
         w3b + (long long)i * 1024, hbuf, ada + 5120, 6144,
         hbuf, 0, 1024, 4096, 1024, 2730, 1.f, 0, 1);
  }

  // ----- final layer -----
  gemm(siluc, 0, 1024, fdw, 0, 1024, fdb, nullptr, nullptr, 0,
       fada, 0, 2048, 16, 2048, 1024, 1.f, 0, 1);
  rms_mod_kernel<<<4096, 256, 0, stream>>>(hbuf, fnw, fada + 1024, fada, 2048, xmbuf);
  gemm(xmbuf, 0, 1024, flw, 0, 1024, flb, nullptr, nullptr, 0,
       flin, 0, 768, 4096, 768, 1024, 1.f, 0, 1);
  unpatchify_kernel<<<blocks(16 * 3 * 65536), 256, 0, stream>>>(flin, outp);
}
